// MultiHeadAttention_63634235457857
// MI455X (gfx1250) — matrix operand order repaired and checked by reading, not compiled, not run on hardware
//
#include <hip/hip_runtime.h>
#include <hip/hip_bf16.h>

#define USE_ASYNC_LDS 1   // gfx1250 GLOBAL_LOAD_ASYNC_TO_LDS_B128 staging path

typedef __attribute__((ext_vector_type(16))) __bf16 v16bf;
typedef __attribute__((ext_vector_type(8)))  float  v8f;

#define HEADS   16
#define DMODEL  1024
#define DHEAD   64
#define SEQ     2048
#define BATCH   2
#define MTOT    (BATCH * SEQ)      // 4096 tokens

// LDS strides (elements) chosen for 16-distinct-bank access + 16B alignment
#define KST 72   // K tile row stride  (32 rows x 64 used)
#define VST 40   // V^T tile row stride (64 rows x 32 used)
#define SST 36   // score tile row stride (16 rows x 32 used, f32)
#define PST 40   // P tile row stride   (16 rows x 32 used, bf16)

// ---------------------------------------------------------------- helpers

__device__ __forceinline__ v8f wmma_bf16(v16bf a, v16bf b, v8f c) {
    return __builtin_amdgcn_wmma_f32_16x16x32_bf16(
        false, a, false, b, (short)0, c, false, false);
}

union Frag { v16bf v; __bf16 h[16]; uint4 q[2]; };

#if USE_ASYNC_LDS
__device__ __forceinline__ void async_copy_b128(void* lds_dst, const void* gsrc) {
    const unsigned d = (unsigned)(unsigned long long)lds_dst;       // LDS offset = low 32 bits
    const unsigned long long s = (unsigned long long)gsrc;
    asm volatile("global_load_async_to_lds_b128 %0, %1, off"
                 :: "v"(d), "v"(s) : "memory");
}
__device__ __forceinline__ void wait_async() {
#if __has_builtin(__builtin_amdgcn_s_wait_asynccnt)
    __builtin_amdgcn_s_wait_asynccnt(0);
#else
    asm volatile("s_wait_asynccnt 0x0" ::: "memory");
#endif
}
#endif

// ---------------------------------------------------------------- fp32 -> bf16 (row-major, vectorized)
__global__ void conv_bf16_kernel(const float* __restrict__ X, __bf16* __restrict__ Y) {
    const size_t i = ((size_t)blockIdx.x * blockDim.x + threadIdx.x) * 4;
    const float4 v = *(const float4*)(X + i);
    union { __bf16 h[4]; uint2 u; } p;
    p.h[0] = (__bf16)v.x; p.h[1] = (__bf16)v.y;
    p.h[2] = (__bf16)v.z; p.h[3] = (__bf16)v.w;
    *(uint2*)(Y + i) = p.u;
}

// ---------------------------------------------------------------- W[K][N] fp32 -> Wt[N][K] bf16 (scaled)
__global__ void transpose_conv_kernel(const float* __restrict__ W,
                                      __bf16* __restrict__ Wt, float scale) {
    __shared__ float tile[32][33];
    const int tx = threadIdx.x & 31;
    const int ty = threadIdx.x >> 5;            // 0..7
    const int k0 = blockIdx.x * 32;
    const int n0 = blockIdx.y * 32;
#pragma unroll
    for (int i = 0; i < 32; i += 8)
        tile[ty + i][tx] = W[(size_t)(k0 + ty + i) * DMODEL + n0 + tx];
    __syncthreads();
#pragma unroll
    for (int i = 0; i < 32; i += 8)
        Wt[(size_t)(n0 + ty + i) * DMODEL + k0 + tx] = (__bf16)(tile[tx][ty + i] * scale);
}

// ---------------------------------------------------------------- GEMM + bias (bf16 x bf16^T -> f32)
// OUT_MODE 0: fp32 row-major ; 1: bf16 [B*H][S][Dh] ; 2: bf16 [B*H][Dh][S]
template <int OUT_MODE>
__global__ void gemm_bias_kernel(const __bf16* __restrict__ Ab,
                                 const __bf16* __restrict__ Wt,
                                 const float* __restrict__ bias, float bscale,
                                 void* __restrict__ out) {
    const int lane = threadIdx.x & 31;
    const int wv   = threadIdx.x >> 5;
    const int m0   = blockIdx.x * 16;
    const int n0   = (blockIdx.y * 4 + wv) * 16;
    const int mrow = lane & 15;
    const int half = lane >> 4;
    const int nn   = lane & 15;

    v8f acc;
#pragma unroll
    for (int r = 0; r < 8; ++r) acc[r] = 0.0f;

    const __bf16* ap = Ab + (size_t)(m0 + mrow) * DMODEL + 16 * half;
    const __bf16* bp = Wt + (size_t)(n0 + nn) * DMODEL + 16 * half;

#pragma unroll 2
    for (int kb = 0; kb < DMODEL; kb += 32) {
        Frag a, b;
        a.q[0] = *(const uint4*)(ap + kb);
        a.q[1] = *(const uint4*)(ap + kb + 8);
        b.q[0] = *(const uint4*)(bp + kb);
        b.q[1] = *(const uint4*)(bp + kb + 8);
        acc = wmma_bf16(a.v, b.v, acc);
    }

    const float bn = bias[n0 + nn] * bscale;

#pragma unroll
    for (int r = 0; r < 8; ++r) {
        const int m = r + 8 * half;
        const float val = acc[r] + bn;
        if (OUT_MODE == 0) {
            ((float*)out)[(size_t)(m0 + m) * DMODEL + n0 + nn] = val;
        } else {
            __bf16* O = (__bf16*)out;
            const int s  = m0 + m;
            const int bb = s >> 11;
            const int sl = s & 2047;
            const int j  = n0 + nn;
            const int h  = j >> 6;
            const int dh = j & 63;
            if (OUT_MODE == 1)
                O[((size_t)(bb * HEADS + h) * SEQ + sl) * DHEAD + dh] = (__bf16)val;
            else
                O[((size_t)(bb * HEADS + h) * DHEAD + dh) * SEQ + sl] = (__bf16)val;
        }
    }
}

// ---------------------------------------------------------------- flash attention
// Q(pre-scaled),K: [B*H][S][Dh] bf16 ; Vt: [B*H][Dh][S] bf16 ; ctx: [B*S][DMODEL] bf16
// 4 waves/block share one (b,h): K/V tiles staged in LDS (double buffered, async).
__global__ void attn_kernel(const __bf16* __restrict__ Qb,
                            const __bf16* __restrict__ Kb,
                            const __bf16* __restrict__ Vt,
                            __bf16* __restrict__ ctx) {
    __shared__ __align__(16) __bf16 Kt[2][32 * KST];   // ~9.2 KB
    __shared__ __align__(16) __bf16 Vs[2][64 * VST];   // ~10.2 KB
    __shared__ __align__(16) float  Sb[4][16 * SST];   // ~9.2 KB
    __shared__ __align__(16) __bf16 Pb[4][16 * PST];   // ~5.1 KB
    __shared__ __align__(16) float  Xb[4][16];

    const int tid  = threadIdx.x;
    const int lane = tid & 31;
    const int wv   = tid >> 5;
    const int bh   = blockIdx.x;
    const int b    = bh >> 4;
    const int h    = bh & 15;
    const int qbase = (blockIdx.y * 4 + wv) * 16;
    const int mrow = lane & 15;
    const int half = lane >> 4;
    const int nn   = lane & 15;

    const __bf16* Kg = Kb + (size_t)bh * SEQ * DHEAD;
    const __bf16* Vg = Vt + (size_t)bh * DHEAD * SEQ;

    // staging piece indices (256 x 16B pieces per tile, 128 threads, 2 passes)
    const int ki0 = tid,        ki1 = tid + 128;
    const int kr0 = ki0 >> 3,   ks0 = ki0 & 7;    // K: row, 16B-chunk
    const int kr1 = ki1 >> 3,   ks1 = ki1 & 7;
    const int vr0 = ki0 >> 2,   vs0 = ki0 & 3;    // V: row, 16B-chunk
    const int vr1 = ki1 >> 2,   vs1 = ki1 & 3;

    // Q fragments (Dh=64 -> two K=32 slabs); 1/sqrt(Dh) pre-folded into Wq/bq
    Frag qa[2];
    {
        const __bf16* qp = Qb + ((size_t)bh * SEQ + qbase + mrow) * DHEAD;
#pragma unroll
        for (int f = 0; f < 2; ++f) {
            qa[f].q[0] = *(const uint4*)(qp + f * 32 + 16 * half);
            qa[f].q[1] = *(const uint4*)(qp + f * 32 + 16 * half + 8);
        }
    }

    v8f o[4];
#pragma unroll
    for (int j = 0; j < 4; ++j)
#pragma unroll
        for (int r = 0; r < 8; ++r) o[j][r] = 0.0f;

    float mrun = -3.0e38f;   // row = lane&15 (duplicated across halves)
    float lrun = 0.0f;

    // ---- stage tile 0
#if USE_ASYNC_LDS
    async_copy_b128(&Kt[0][kr0 * KST + ks0 * 8], Kg + (size_t)kr0 * DHEAD + ks0 * 8);
    async_copy_b128(&Kt[0][kr1 * KST + ks1 * 8], Kg + (size_t)kr1 * DHEAD + ks1 * 8);
    async_copy_b128(&Vs[0][vr0 * VST + vs0 * 8], Vg + (size_t)vr0 * SEQ + vs0 * 8);
    async_copy_b128(&Vs[0][vr1 * VST + vs1 * 8], Vg + (size_t)vr1 * SEQ + vs1 * 8);
    wait_async();
#else
    *(uint4*)&Kt[0][kr0 * KST + ks0 * 8] = *(const uint4*)(Kg + (size_t)kr0 * DHEAD + ks0 * 8);
    *(uint4*)&Kt[0][kr1 * KST + ks1 * 8] = *(const uint4*)(Kg + (size_t)kr1 * DHEAD + ks1 * 8);
    *(uint4*)&Vs[0][vr0 * VST + vs0 * 8] = *(const uint4*)(Vg + (size_t)vr0 * SEQ + vs0 * 8);
    *(uint4*)&Vs[0][vr1 * VST + vs1 * 8] = *(const uint4*)(Vg + (size_t)vr1 * SEQ + vs1 * 8);
#endif
    __syncthreads();

    for (int kb = 0; kb < SEQ / 32; ++kb) {
        const int kv0 = kb * 32;
        const int cur = kb & 1;
        const int nxt = cur ^ 1;
        const bool more = (kb + 1) < (SEQ / 32);

        // ---- stage next tile (nxt buffer is free: its readers hit the kb-1 barrier)
#if USE_ASYNC_LDS
        if (more) {
            const int kv1 = kv0 + 32;
            async_copy_b128(&Kt[nxt][kr0 * KST + ks0 * 8], Kg + (size_t)(kv1 + kr0) * DHEAD + ks0 * 8);
            async_copy_b128(&Kt[nxt][kr1 * KST + ks1 * 8], Kg + (size_t)(kv1 + kr1) * DHEAD + ks1 * 8);
            async_copy_b128(&Vs[nxt][vr0 * VST + vs0 * 8], Vg + (size_t)vr0 * SEQ + kv1 + vs0 * 8);
            async_copy_b128(&Vs[nxt][vr1 * VST + vs1 * 8], Vg + (size_t)vr1 * SEQ + kv1 + vs1 * 8);
            __builtin_prefetch((const void*)(Kg + (size_t)(kv1 + 32) * DHEAD), 0, 1);
        }
#else
        uint4 kreg0, kreg1, vreg0, vreg1;
        if (more) {
            const int kv1 = kv0 + 32;
            kreg0 = *(const uint4*)(Kg + (size_t)(kv1 + kr0) * DHEAD + ks0 * 8);
            kreg1 = *(const uint4*)(Kg + (size_t)(kv1 + kr1) * DHEAD + ks1 * 8);
            vreg0 = *(const uint4*)(Vg + (size_t)vr0 * SEQ + kv1 + vs0 * 8);
            vreg1 = *(const uint4*)(Vg + (size_t)vr1 * SEQ + kv1 + vs1 * 8);
            __builtin_prefetch((const void*)(Kg + (size_t)(kv1 + 32) * DHEAD), 0, 1);
        }
#endif

        // ---- scores: load 4 K fragments from LDS, then 4 WMMAs
        Frag kf[2][2];
#pragma unroll
        for (int t = 0; t < 2; ++t)
#pragma unroll
            for (int f = 0; f < 2; ++f) {
                const __bf16* kr = &Kt[cur][(t * 16 + nn) * KST + f * 32 + 16 * half];
                kf[t][f].q[0] = *(const uint4*)(kr);
                kf[t][f].q[1] = *(const uint4*)(kr + 8);
            }
        v8f s0, s1;
#pragma unroll
        for (int r = 0; r < 8; ++r) { s0[r] = 0.0f; s1[r] = 0.0f; }
        s0 = wmma_bf16(qa[0].v, kf[0][0].v, s0);
        s1 = wmma_bf16(qa[0].v, kf[1][0].v, s1);
        s0 = wmma_bf16(qa[1].v, kf[0][1].v, s0);
        s1 = wmma_bf16(qa[1].v, kf[1][1].v, s1);

        // ---- dump C-layout scores to per-wave LDS tile
        float* sb = Sb[wv];
#pragma unroll
        for (int r = 0; r < 8; ++r) {
            const int m = 8 * half + r;
            sb[m * SST + nn]      = s0[r];
            sb[m * SST + 16 + nn] = s1[r];
        }
        asm volatile("" ::: "memory");   // intra-wave LDS ordering only

        // ---- transposed softmax: lane owns row (lane&15), columns half*16..+15
        const float* srow = &sb[mrow * SST + half * 16];
        const float4 x0 = *(const float4*)(srow + 0);
        const float4 x1 = *(const float4*)(srow + 4);
        const float4 x2 = *(const float4*)(srow + 8);
        const float4 x3 = *(const float4*)(srow + 12);

        float mx = fmaxf(fmaxf(fmaxf(x0.x, x0.y), fmaxf(x0.z, x0.w)),
                         fmaxf(fmaxf(x1.x, x1.y), fmaxf(x1.z, x1.w)));
        mx = fmaxf(mx, fmaxf(fmaxf(fmaxf(x2.x, x2.y), fmaxf(x2.z, x2.w)),
                             fmaxf(fmaxf(x3.x, x3.y), fmaxf(x3.z, x3.w))));
        mx = fmaxf(mx, __shfl_xor(mx, 16, 32));

        const float mnew  = fmaxf(mrun, mx);
        const float alpha = __expf(mrun - mnew);
        mrun = mnew;

        float p[16];
        p[0]  = __expf(x0.x - mnew); p[1]  = __expf(x0.y - mnew);
        p[2]  = __expf(x0.z - mnew); p[3]  = __expf(x0.w - mnew);
        p[4]  = __expf(x1.x - mnew); p[5]  = __expf(x1.y - mnew);
        p[6]  = __expf(x1.z - mnew); p[7]  = __expf(x1.w - mnew);
        p[8]  = __expf(x2.x - mnew); p[9]  = __expf(x2.y - mnew);
        p[10] = __expf(x2.z - mnew); p[11] = __expf(x2.w - mnew);
        p[12] = __expf(x3.x - mnew); p[13] = __expf(x3.y - mnew);
        p[14] = __expf(x3.z - mnew); p[15] = __expf(x3.w - mnew);

        float sum = 0.0f;
#pragma unroll
        for (int i = 0; i < 16; ++i) sum += p[i];
        sum += __shfl_xor(sum, 16, 32);
        lrun = lrun * alpha + sum;

        // ---- write P (bf16 tile, 32 contiguous bytes per lane)
        Frag pk;
#pragma unroll
        for (int i = 0; i < 16; ++i) pk.h[i] = (__bf16)p[i];
        __bf16* pw = &Pb[wv][mrow * PST + half * 16];
        *(uint4*)(pw)     = pk.q[0];
        *(uint4*)(pw + 8) = pk.q[1];

        // ---- broadcast per-row alpha, rescale O (C layout rows 8*half+r)
        if (half == 0) Xb[wv][mrow] = alpha;
        asm volatile("" ::: "memory");
        const float4 a0 = *(const float4*)&Xb[wv][8 * half];
        const float4 a1 = *(const float4*)&Xb[wv][8 * half + 4];
        const float alr[8] = {a0.x, a0.y, a0.z, a0.w, a1.x, a1.y, a1.z, a1.w};
#pragma unroll
        for (int j = 0; j < 4; ++j)
#pragma unroll
            for (int r = 0; r < 8; ++r) o[j][r] *= alr[r];

        // ---- A-fragment of P, 4 V fragments from LDS, 4 WMMAs
        Frag pa;
        const __bf16* pl = &Pb[wv][mrow * PST];
        pa.q[0] = *(const uint4*)(pl + 16 * half);
        pa.q[1] = *(const uint4*)(pl + 16 * half + 8);
        Frag vf[4];
#pragma unroll
        for (int j = 0; j < 4; ++j) {
            const __bf16* vr = &Vs[cur][(j * 16 + nn) * VST + 16 * half];
            vf[j].q[0] = *(const uint4*)(vr);
            vf[j].q[1] = *(const uint4*)(vr + 8);
        }
#pragma unroll
        for (int j = 0; j < 4; ++j)
            o[j] = wmma_bf16(pa.v, vf[j].v, o[j]);

        // ---- complete staging, block-wide handoff barrier
#if USE_ASYNC_LDS
        wait_async();
#else
        if (more) {
            *(uint4*)&Kt[nxt][kr0 * KST + ks0 * 8] = kreg0;
            *(uint4*)&Kt[nxt][kr1 * KST + ks1 * 8] = kreg1;
            *(uint4*)&Vs[nxt][vr0 * VST + vs0 * 8] = vreg0;
            *(uint4*)&Vs[nxt][vr1 * VST + vs1 * 8] = vreg1;
        }
#endif
        __syncthreads();
    }

    // ---- broadcast 1/l, normalize, emit ctx
    if (half == 0) Xb[wv][mrow] = 1.0f / lrun;
    asm volatile("" ::: "memory");
    const float4 i0 = *(const float4*)&Xb[wv][8 * half];
    const float4 i1 = *(const float4*)&Xb[wv][8 * half + 4];
    const float inv[8] = {i0.x, i0.y, i0.z, i0.w, i1.x, i1.y, i1.z, i1.w};
#pragma unroll
    for (int r = 0; r < 8; ++r) {
        const int sl = qbase + 8 * half + r;
        const size_t row = (size_t)(b * SEQ + sl) * DMODEL + h * DHEAD;
#pragma unroll
        for (int j = 0; j < 4; ++j)
            ctx[row + j * 16 + nn] = (__bf16)(o[j][r] * inv[r]);
    }
}

// ---------------------------------------------------------------- launcher

extern "C" void kernel_launch(void* const* d_in, const int* in_sizes, int n_in,
                              void* d_out, int out_size, void* d_ws, size_t ws_size,
                              hipStream_t stream) {
    (void)in_sizes; (void)n_in; (void)out_size; (void)ws_size;
    const float* key   = (const float*)d_in[0];
    const float* value = (const float*)d_in[1];
    const float* query = (const float*)d_in[2];
    const float* Wk = (const float*)d_in[3];
    const float* bk = (const float*)d_in[4];
    const float* Wv = (const float*)d_in[5];
    const float* bv = (const float*)d_in[6];
    const float* Wq = (const float*)d_in[7];
    const float* bq = (const float*)d_in[8];
    const float* Wo = (const float*)d_in[9];
    const float* bo = (const float*)d_in[10];

    char* ws = (char*)d_ws;
    const size_t ASZ = (size_t)MTOT * DMODEL * sizeof(__bf16);     // 8 MB
    __bf16* Qb  = (__bf16*)(ws + 0 * ASZ);
    __bf16* Kbf = (__bf16*)(ws + 1 * ASZ);
    __bf16* Vt  = (__bf16*)(ws + 2 * ASZ);
    __bf16* ctx = (__bf16*)(ws + 3 * ASZ);
    __bf16* Xbf = (__bf16*)(ws + 4 * ASZ);                         // reused per input
    __bf16* Wt  = (__bf16*)(ws + 5 * ASZ);                         // reused per weight

    const dim3 blk(128);
    const dim3 gconv((MTOT * DMODEL / 4) / 256);
    const dim3 gtr(DMODEL / 32, DMODEL / 32);
    const dim3 gproj(MTOT / 16, DMODEL / 64);
    const dim3 gattn(BATCH * HEADS, (SEQ / 16) / 4);

    const float qscale = 0.125f;  // 1/sqrt(DHEAD), folded into Wq/bq

    conv_bf16_kernel<<<gconv, 256, 0, stream>>>(key, Xbf);
    transpose_conv_kernel<<<gtr, 256, 0, stream>>>(Wk, Wt, 1.0f);
    gemm_bias_kernel<1><<<gproj, blk, 0, stream>>>(Xbf, Wt, bk, 1.0f, (void*)Kbf);

    conv_bf16_kernel<<<gconv, 256, 0, stream>>>(value, Xbf);
    transpose_conv_kernel<<<gtr, 256, 0, stream>>>(Wv, Wt, 1.0f);
    gemm_bias_kernel<2><<<gproj, blk, 0, stream>>>(Xbf, Wt, bv, 1.0f, (void*)Vt);

    conv_bf16_kernel<<<gconv, 256, 0, stream>>>(query, Xbf);
    transpose_conv_kernel<<<gtr, 256, 0, stream>>>(Wq, Wt, qscale);
    gemm_bias_kernel<1><<<gproj, blk, 0, stream>>>(Xbf, Wt, bq, qscale, (void*)Qb);

    attn_kernel<<<gattn, blk, 0, stream>>>(Qb, Kbf, Vt, ctx);

    transpose_conv_kernel<<<gtr, 256, 0, stream>>>(Wo, Wt, 1.0f);
    gemm_bias_kernel<0><<<gproj, blk, 0, stream>>>(ctx, Wt, bo, 1.0f, d_out);
}
